// SAGIN_89318139887642
// MI455X (gfx1250) — matrix operand order repaired and checked by reading, not compiled, not run on hardware
//
#include <hip/hip_runtime.h>
#include <hip/hip_bf16.h>

typedef __attribute__((ext_vector_type(16))) _Float16 v16h;
typedef __attribute__((ext_vector_type(8)))  _Float16 v8h;
typedef __attribute__((ext_vector_type(8)))  float    v8f;

#define TPB 256

__device__ __forceinline__ float elu1(float v) {
  return v > 0.0f ? v : expm1f(v);
}

// ---------------- elementwise / setup kernels ----------------

__global__ void k_zero(float* __restrict__ p, long long n) {
  long long i = (long long)blockIdx.x * blockDim.x + threadIdx.x;
  if (i < n) p[i] = 0.0f;
}

__global__ void k_deg(const int* __restrict__ dst, float* __restrict__ deg, int E) {
  int i = blockIdx.x * blockDim.x + threadIdx.x;
  if (i < E) atomicAdd(&deg[dst[i]], 1.0f);
}

__global__ void k_deg_finalize(float* __restrict__ deg, int n) {
  int i = blockIdx.x * blockDim.x + threadIdx.x;
  if (i < n) { float d = deg[i]; deg[i] = 1.0f / fmaxf(d, 1.0f); }
}

// mean-aggregation scatter: agg[dst] += h[src] (divide by deg later)
__global__ void k_scatter(const float* __restrict__ h, const int* __restrict__ src,
                          const int* __restrict__ dst, float* __restrict__ agg,
                          long long total, int d) {
  long long i = (long long)blockIdx.x * blockDim.x + threadIdx.x;
  if (i >= total) return;
  int e = (int)(i / d);
  int f = (int)(i % d);
  atomicAdd(&agg[(long long)dst[e] * d + f], h[(long long)src[e] * d + f]);
}

// X2 = f16([agg * deg_inv , h])  (N x 2d), row-major
__global__ void k_pack2(const float* __restrict__ agg, const float* __restrict__ h,
                        const float* __restrict__ deg_inv, _Float16* __restrict__ X2,
                        long long total, int d) {
  long long i = (long long)blockIdx.x * blockDim.x + threadIdx.x;
  if (i >= total) return;
  int K2 = 2 * d;
  long long r = i / K2;
  int k = (int)(i % K2);
  float v = (k < d) ? agg[r * d + k] * deg_inv[r] : h[r * d + (k - d)];
  X2[i] = (_Float16)v;
}

__global__ void k_pack1(const float* __restrict__ h, _Float16* __restrict__ X2, long long n) {
  long long i = (long long)blockIdx.x * blockDim.x + threadIdx.x;
  if (i < n) X2[i] = (_Float16)h[i];
}

// Pack the (virtual) stacked weight [Wl ; Wr] (K x dout, K = dsplit + rest) directly
// into the WMMA B-fragment layout:
//   Wpk[ ((kt*tilesN + nt)*32 + lane)*16 + e ] = W[(kt*32 + (lane>>4)*8 + (e&7) + (e>>3)*16)*dout + nt*16 + (lane&15)]
// so each GEMM lane loads its 16 halves with two contiguous b128 loads.
__global__ void k_packB(const float* __restrict__ Wl, const float* __restrict__ Wr,
                        _Float16* __restrict__ Wpk, int dsplit, int K, int dout) {
  int i = blockIdx.x * blockDim.x + threadIdx.x;
  int total = K * dout;
  if (i >= total) return;
  const int tilesN = dout >> 4;
  int frag = i >> 9;           // 512 halves per 32x16 fragment
  int r    = i & 511;
  int lane = r >> 4;
  int e    = r & 15;
  int kt = frag / tilesN;
  int nt = frag % tilesN;
  int k = kt * 32 + (lane >> 4) * 8 + (e & 7) + (e >> 3) * 16;
  int n = nt * 16 + (lane & 15);
  float v = (k < dsplit) ? Wl[k * dout + n] : Wr[(k - dsplit) * dout + n];
  Wpk[i] = (_Float16)v;
}

// ---------------- WMMA GEMM: Y(f32) = X(f16, N x K) @ W(prepacked frags) ----------------
// one wave32 per 16x(16*TN) output strip; K multiple of 32; N multiple of 16.
template <int TN>
__global__ void k_gemm_wmma(const _Float16* __restrict__ X, const _Float16* __restrict__ Wpk,
                            float* __restrict__ Y, int nrows, int K, int dout) {
  const int lane   = threadIdx.x & 31;
  const int wave   = threadIdx.x >> 5;
  const int tilesN = dout >> 4;
  const int groups = tilesN / TN;
  const long long tile = (long long)blockIdx.x * (blockDim.x >> 5) + wave;
  const long long totalTiles = (long long)(nrows >> 4) * groups;
  if (tile >= totalTiles) return;  // wave-uniform exit: EXEC all-ones inside
  const int tm  = (int)(tile / groups);
  const int tn0 = (int)(tile % groups) * TN;
  const int lrow  = lane & 15;
  const int khalf = lane >> 4;
  const long long arow = (long long)(tm * 16 + lrow);

  v8f acc[TN];
#pragma unroll
  for (int t = 0; t < TN; ++t) acc[t] = (v8f){};

  for (int k0 = 0; k0 < K; k0 += 32) {
    // A fragment: lane(0..15)=row M, khalf selects K sub-bands {0..7,16..23} / {8..15,24..31}
    const _Float16* ap = X + arow * K + k0 + khalf * 8;
    v8h alo = *(const v8h*)(ap);
    v8h ahi = *(const v8h*)(ap + 16);
    v16h a;
#pragma unroll
    for (int j = 0; j < 8; ++j) { a[j] = alo[j]; a[8 + j] = ahi[j]; }

#pragma unroll
    for (int t = 0; t < TN; ++t) {
      // B fragment: prepacked, contiguous 16 halves per lane
      const _Float16* bp = Wpk + ((((long long)(k0 >> 5) * tilesN) + (tn0 + t)) << 9)
                               + (lane << 4);
      v8h blo = *(const v8h*)(bp);
      v8h bhi = *(const v8h*)(bp + 8);
      v16h b;
#pragma unroll
      for (int j = 0; j < 8; ++j) { b[j] = blo[j]; b[8 + j] = bhi[j]; }

      acc[t] = __builtin_amdgcn_wmma_f32_16x16x32_f16(
          /*neg_a=*/false, a, /*neg_b=*/false, b,
          /*c_mod=*/(short)0, acc[t], /*reuse_a=*/false, /*reuse_b=*/false);
    }
  }

  // C layout: VGPR r -> row tm*16 + khalf*8 + r, col = lane&15 within tile
  const int rbase = tm * 16 + khalf * 8;
#pragma unroll
  for (int t = 0; t < TN; ++t) {
    const int col = (tn0 + t) * 16 + lrow;
#pragma unroll
    for (int r = 0; r < 8; ++r)
      Y[(long long)(rbase + r) * dout + col] = acc[t][r];
  }
}

// ---------------- BatchNorm stats / apply ----------------

__global__ void k_bnstats(const float* __restrict__ Y, float* __restrict__ stats,
                          int nrows, int dout) {
  // dout is a power of two (16..256); blockDim = 256
  int c   = threadIdx.x & (dout - 1);
  int per = blockDim.x / dout;
  int sub = threadIdx.x / dout;
  float s = 0.0f, s2 = 0.0f;
  for (long long r = (long long)blockIdx.x * per + sub; r < nrows;
       r += (long long)gridDim.x * per) {
    float v = Y[r * dout + c];
    s += v; s2 += v * v;
  }
  atomicAdd(&stats[c], s);
  atomicAdd(&stats[dout + c], s2);
}

__global__ void k_bn_elu(const float* __restrict__ Y, const float* __restrict__ stats,
                         const float* __restrict__ g, const float* __restrict__ bb,
                         float* __restrict__ hnext, long long total, int dout,
                         int skipDim, float invN) {
  long long i = (long long)blockIdx.x * blockDim.x + threadIdx.x;
  if (i >= total) return;
  long long r = i / dout;
  int c = (int)(i % dout);
  float mean = stats[c] * invN;
  float var  = stats[dout + c] * invN - mean * mean;
  float rs   = rsqrtf(var + 1e-5f);
  float v    = g[c] * (Y[i] - mean) * rs + bb[c];
  hnext[r * (long long)(skipDim + dout) + skipDim + c] = elu1(v);
}

// ---------------- skip path / misc ----------------

__global__ void k_copycols(const float* __restrict__ hsrc, float* __restrict__ hdst,
                           long long total, int dsrc, int ld) {
  long long i = (long long)blockIdx.x * blockDim.x + threadIdx.x;
  if (i >= total) return;
  long long r = i / dsrc;
  int c = (int)(i % dsrc);
  hdst[r * (long long)ld + c] = hsrc[i];
}

// layer-0 skip: (N x 32) @ (32 x 16) into columns [0,16) of hnext
__global__ void k_skipproj(const float* __restrict__ x, const float* __restrict__ W,
                           float* __restrict__ hdst, int nrows, int ld) {
  long long i = (long long)blockIdx.x * blockDim.x + threadIdx.x;
  if (i >= (long long)nrows * 16) return;
  long long r = i / 16;
  int c = (int)(i % 16);
  float s = 0.0f;
#pragma unroll
  for (int k = 0; k < 32; ++k) s += x[r * 32 + k] * W[k * 16 + c];
  hdst[r * (long long)ld + c] = s;
}

__global__ void k_elu_copy(const float* __restrict__ Y, float* __restrict__ hdst, long long n) {
  long long i = (long long)blockIdx.x * blockDim.x + threadIdx.x;
  if (i < n) hdst[i] = elu1(Y[i]);
}

// ---------------- pooling + MLP head ----------------

__global__ void k_pool(const float* __restrict__ h, const int* __restrict__ batch,
                       float* __restrict__ hg, long long total) {
  long long i = (long long)blockIdx.x * blockDim.x + threadIdx.x;
  if (i >= total) return;
  long long r = i / 512;
  int c = (int)(i % 512);
  atomicAdd(&hg[(long long)batch[r] * 512 + c], h[i]);
}

__global__ void k_count(const int* __restrict__ batch, float* __restrict__ cnt, int n) {
  int i = blockIdx.x * blockDim.x + threadIdx.x;
  if (i < n) atomicAdd(&cnt[batch[i]], 1.0f);
}

__global__ void k_head(const float* __restrict__ hg, const float* __restrict__ cnt,
                       const float* __restrict__ eFeat,
                       const float* __restrict__ fc1_W, const float* __restrict__ fc1_b,
                       const float* __restrict__ fc3_W, const float* __restrict__ fc3_b,
                       float* __restrict__ out) {
  __shared__ float vec[528];
  __shared__ float h1[512];
  int g = blockIdx.x;
  float inv = 1.0f / fmaxf(cnt[g], 1.0f);
  for (int c = threadIdx.x; c < 512; c += blockDim.x) vec[c] = hg[(long long)g * 512 + c] * inv;
  for (int c = threadIdx.x; c < 16; c += blockDim.x)  vec[512 + c] = eFeat[g * 16 + c];
  __syncthreads();
  for (int o = threadIdx.x; o < 512; o += blockDim.x) {
    float s = fc1_b[o];
    for (int k = 0; k < 528; ++k) s += vec[k] * fc1_W[k * 512 + o];
    h1[o] = elu1(s);
  }
  __syncthreads();
  for (int o = threadIdx.x; o < 8; o += blockDim.x) {
    float s = fc3_b[o];
    for (int k = 0; k < 512; ++k) s += h1[k] * fc3_W[k * 8 + o];
    out[g * 8 + o] = s;
  }
}

// ---------------- host orchestration ----------------

static inline int nblk(long long n) { return (int)((n + TPB - 1) / TPB); }

static inline void launch_gemm(const _Float16* X, const _Float16* Wpk, float* Y,
                               int N, int K, int dout, hipStream_t stream) {
  const int tilesN = dout / 16;
  if ((tilesN & 1) == 0) {
    long long tiles = (long long)(N / 16) * (tilesN / 2);
    k_gemm_wmma<2><<<(int)((tiles + 7) / 8), TPB, 0, stream>>>(X, Wpk, Y, N, K, dout);
  } else {
    long long tiles = (long long)(N / 16) * tilesN;
    k_gemm_wmma<1><<<(int)((tiles + 7) / 8), TPB, 0, stream>>>(X, Wpk, Y, N, K, dout);
  }
}

extern "C" void kernel_launch(void* const* d_in, const int* in_sizes, int n_in,
                              void* d_out, int out_size, void* d_ws, size_t ws_size,
                              hipStream_t stream) {
  (void)n_in; (void)out_size; (void)ws_size;

  const int N = in_sizes[0] / 32;
  const int E = in_sizes[1] / 2;
  const int B = in_sizes[3] / 16;

  const float* x     = (const float*)d_in[0];
  const int*   ei    = (const int*)d_in[1];
  const int*   src   = ei;
  const int*   dst   = ei + E;
  const int*   batch = (const int*)d_in[2];
  const float* eFeat = (const float*)d_in[3];

  const float* Wl[7]; const float* Wr[7];
  for (int i = 0; i < 7; ++i) { Wl[i] = (const float*)d_in[4 + 3 * i];
                                Wr[i] = (const float*)d_in[5 + 3 * i]; }
  const float* ng[7]; const float* nb[7];
  for (int i = 0; i < 7; ++i) { ng[i] = (const float*)d_in[25 + 2 * i];
                                nb[i] = (const float*)d_in[26 + 2 * i]; }
  const float* skip1_W = (const float*)d_in[39];
  const float* red_W   = (const float*)d_in[40];
  const float* fc1_W   = (const float*)d_in[41];
  const float* fc1_b   = (const float*)d_in[42];
  const float* fc3_W   = (const float*)d_in[43];
  const float* fc3_b   = (const float*)d_in[44];

  // workspace layout (all f32 unless noted); base is 256B aligned from hipMalloc
  float*     deg_inv = (float*)d_ws;                              // N
  float*     agg     = deg_inv + (((long long)N + 255) & ~255LL); // N*256
  _Float16*  X2      = (_Float16*)(agg + (long long)N * 256);     // N*512 halves
  float*     Yb      = (float*)(X2 + (long long)N * 512);         // N*256
  float*     hA      = Yb + (long long)N * 256;                   // N*512
  float*     hB      = hA + (long long)N * 512;                   // N*512
  _Float16*  Wpk     = (_Float16*)(hB + (long long)N * 512);      // 512*256 halves
  float*     stats   = (float*)(Wpk + 512 * 256);                 // 512
  float*     hg      = stats + 512;                               // B*512
  float*     cnt     = hg + (long long)B * 512;                   // B

  static const int DIN[7]  = {32, 32, 64, 32, 64, 128, 256};
  static const int DOUT[7] = {16, 32, 64, 32, 64, 128, 256};
  const float invN = 1.0f / (float)N;

  // degree -> reciprocal (computed once)
  k_zero<<<nblk(N), TPB, 0, stream>>>(deg_inv, N);
  k_deg<<<nblk(E), TPB, 0, stream>>>(dst, deg_inv, E);
  k_deg_finalize<<<nblk(N), TPB, 0, stream>>>(deg_inv, N);

  const float* hcur = x;
  float* bufs[2] = {hA, hB};
  int cur = 0;

  for (int i = 0; i < 7; ++i) {
    const int d = DIN[i];
    const int dout = DOUT[i];
    const int skipDim = (i == 0) ? 16 : d;
    const int ld = skipDim + dout;
    float* hnext = bufs[cur];

    // mean aggregation
    k_zero<<<nblk((long long)N * d), TPB, 0, stream>>>(agg, (long long)N * d);
    k_scatter<<<nblk((long long)E * d), TPB, 0, stream>>>(hcur, src, dst, agg,
                                                          (long long)E * d, d);
    // pack fused operand [agg/deg, h] and fragment-packed [Wl;Wr], then WMMA GEMM
    k_pack2<<<nblk((long long)N * 2 * d), TPB, 0, stream>>>(agg, hcur, deg_inv, X2,
                                                            (long long)N * 2 * d, d);
    k_packB<<<nblk(2 * d * dout), TPB, 0, stream>>>(Wl[i], Wr[i], Wpk, d, 2 * d, dout);
    launch_gemm(X2, Wpk, Yb, N, 2 * d, dout, stream);

    // BN stats (conv bias cancels under BN)
    k_zero<<<nblk(2 * dout), TPB, 0, stream>>>(stats, 2 * dout);
    k_bnstats<<<512, TPB, 0, stream>>>(Yb, stats, N, dout);
    // skip columns
    if (i == 0)
      k_skipproj<<<nblk((long long)N * 16), TPB, 0, stream>>>(x, skip1_W, hnext, N, ld);
    else
      k_copycols<<<nblk((long long)N * d), TPB, 0, stream>>>(hcur, hnext,
                                                             (long long)N * d, d, ld);
    // BN + ELU into columns [skipDim, skipDim+dout)
    k_bn_elu<<<nblk((long long)N * dout), TPB, 0, stream>>>(Yb, stats, ng[i], nb[i], hnext,
                                                            (long long)N * dout, dout,
                                                            skipDim, invN);
    hcur = hnext;
    cur ^= 1;

    if (i == 2) {
      // redconv: elu(h(128) @ red_W(128x32)), no bias/BN
      float* hred = bufs[cur];
      k_pack1<<<nblk((long long)N * 128), TPB, 0, stream>>>(hcur, X2, (long long)N * 128);
      k_packB<<<nblk(128 * 32), TPB, 0, stream>>>(red_W, red_W, Wpk, 128, 128, 32);
      launch_gemm(X2, Wpk, Yb, N, 128, 32, stream);
      k_elu_copy<<<nblk((long long)N * 32), TPB, 0, stream>>>(Yb, hred, (long long)N * 32);
      hcur = hred;
      cur ^= 1;
    }
  }

  // global mean pool (final h is N x 512) + MLP head
  k_zero<<<nblk((long long)B * 512 + B), TPB, 0, stream>>>(hg, (long long)B * 512 + B);
  k_pool<<<nblk((long long)N * 512), TPB, 0, stream>>>(hcur, batch, hg, (long long)N * 512);
  k_count<<<nblk(N), TPB, 0, stream>>>(batch, cnt, N);
  k_head<<<B, 128, 0, stream>>>(hg, cnt, eFeat, fc1_W, fc1_b, fc3_W, fc3_b, (float*)d_out);
}
